// LinearTransformerLayer_78237124263954
// MI455X (gfx1250) — matrix operand order repaired and checked by reading, not compiled, not run on hardware
//
#include <hip/hip_runtime.h>
#include <math.h>

typedef unsigned short u16t;
typedef __attribute__((ext_vector_type(8)))  u16t   u16x8;
typedef __attribute__((ext_vector_type(16))) u16t   u16x16;
typedef __attribute__((ext_vector_type(16))) __bf16 v16bf;
typedef __attribute__((ext_vector_type(8)))  float  v8f;

// fp32 -> bf16 round-to-nearest-even
__device__ __forceinline__ u16t f2bf(float f) {
    unsigned u = __builtin_bit_cast(unsigned, f);
    unsigned r = (u + 0x7FFFu + ((u >> 16) & 1u)) >> 16;
    return (u16t)r;
}

__device__ __forceinline__ v16bf make_frag(u16x8 lo, u16x8 hi) {
    u16x16 t = __builtin_shufflevector(lo, hi, 0,1,2,3,4,5,6,7,8,9,10,11,12,13,14,15);
    return __builtin_bit_cast(v16bf, t);
}

__device__ __forceinline__ v8f wmma_bf16(v16bf a, v16bf b, v8f c) {
    // D(f32 16x16) = A(16x32 bf16) * B(32x16 bf16) + C
    return __builtin_amdgcn_wmma_f32_16x16x32_bf16(false, a, false, b, (short)0, c, false, false);
}

// ---------------------------------------------------------------------------
// fp32 -> bf16 bulk convert
__global__ void cvt_f32_bf16(const float* __restrict__ src, u16t* __restrict__ dst, int n) {
    int i = blockIdx.x * 256 + threadIdx.x;
    if (i < n) dst[i] = f2bf(src[i]);
}

// ---------------------------------------------------------------------------
// out[m,n] = sum_k A[m,k] * W[n,k] + bias[n]   (A:[Mx1024] bf16, W:[1024x1024] bf16 N-major)
// mode 0: phi(.) -> bf16 row-major           (Q / K projections)
// mode 2: (.)    -> bf16 transposed [b][n][t] (V projection -> Vt)
// mode 3: (.)    -> f32 row-major            (output projection)
// block = 128 threads = 4 waves; wave tile 32x64; block tile 128x64
__global__ __launch_bounds__(128) void proj_kernel(
    const u16t* __restrict__ A, const u16t* __restrict__ W,
    const float* __restrict__ bias,
    u16t* __restrict__ out_bf, float* __restrict__ out_f32, int mode) {
    const int D = 1024;
    const int lane = threadIdx.x & 31;
    const int wave = threadIdx.x >> 5;
    const int half = lane >> 4;
    const int ln   = lane & 15;
    const int row0 = blockIdx.x * 128 + wave * 32;
    const int c0   = blockIdx.y * 64;

    const v8f vzero = {0.f,0.f,0.f,0.f,0.f,0.f,0.f,0.f};
    v8f acc[2][4];
#pragma unroll
    for (int m = 0; m < 2; ++m)
#pragma unroll
        for (int n = 0; n < 4; ++n) acc[m][n] = vzero;

    for (int d0 = 0; d0 < D; d0 += 32) {
        v16bf af[2];
#pragma unroll
        for (int m = 0; m < 2; ++m) {
            const u16t* ap = A + (size_t)(row0 + m*16 + ln) * D + d0;
            af[m] = make_frag(*(const u16x8*)(ap + 8*half),
                              *(const u16x8*)(ap + 16 + 8*half));
        }
#pragma unroll
        for (int n = 0; n < 4; ++n) {
            const u16t* wp = W + (size_t)(c0 + n*16 + ln) * D + d0 + 8*half;
            v16bf bf = make_frag(*(const u16x8*)wp, *(const u16x8*)(wp + 16));
#pragma unroll
            for (int m = 0; m < 2; ++m) acc[m][n] = wmma_bf16(af[m], bf, acc[m][n]);
        }
    }

#pragma unroll
    for (int n = 0; n < 4; ++n) {
        const int col = c0 + n*16 + ln;
        const float bs = bias[col];
#pragma unroll
        for (int m = 0; m < 2; ++m) {
            if (mode == 2) {               // Vt[b][col][t], 8 consecutive t per lane
                int rbase = row0 + m*16 + 8*half;
                int bb = rbase >> 11;      // T = 2048
                int t0 = rbase & 2047;
                u16x8 pk;
#pragma unroll
                for (int v = 0; v < 8; ++v) pk[v] = f2bf(acc[m][n][v] + bs);
                *(u16x8*)(out_bf + (((size_t)(bb*1024 + col)) << 11) + t0) = pk;
            } else {
#pragma unroll
                for (int v = 0; v < 8; ++v) {
                    int row = row0 + m*16 + 8*half + v;
                    float val = acc[m][n][v] + bs;
                    if (mode == 0) {       // phi = elu + 1
                        val = val > 0.f ? val + 1.f : __expf(val);
                        out_bf[(size_t)row * D + col] = f2bf(val);
                    } else {
                        out_f32[(size_t)row * D + col] = val;
                    }
                }
            }
        }
    }
}

// ---------------------------------------------------------------------------
// Flash-style causal linear attention (un-normalized) + denominator + /den.
// grid = (T/32, B, 2); block = 128 (4 waves). Query tile = 32, key step = 32.
// Stage1: A^T = K*Q^T per wave subtile -> mask -> den (ds_add_f32) -> bf16 LDS.
// Stage2: num += A * V using Vt (transposed V), 128 cols per wave.
__global__ __launch_bounds__(128) void attn_kernel(
    const u16t* __restrict__ Qp, const u16t* __restrict__ Kp,
    const u16t* __restrict__ Vt, u16t* __restrict__ Og) {
    const int D = 1024, T = 2048;
    __shared__ u16t As[32][48];     // A^T->A tile, padded rows (96B) for alignment
    __shared__ float den_s[32];

    const int lane = threadIdx.x & 31;
    const int wave = threadIdx.x >> 5;
    const int half = lane >> 4;
    const int ln   = lane & 15;
    const int qt   = blockIdx.x;
    const int b    = blockIdx.y;
    const int colbase = blockIdx.z * 512 + wave * 128;
    const int keyoff  = (wave & 1) * 16;  // stage-1 subtile: rows (keys)
    const int qoff    = (wave >> 1) * 16; // stage-1 subtile: cols (queries)

    if (threadIdx.x < 32) den_s[threadIdx.x] = 0.f;

    const v8f vzero = {0.f,0.f,0.f,0.f,0.f,0.f,0.f,0.f};
    v8f acc[2][8];
#pragma unroll
    for (int m = 0; m < 2; ++m)
#pragma unroll
        for (int n = 0; n < 8; ++n) acc[m][n] = vzero;
    __syncthreads();

    for (int j = 0; j <= qt; ++j) {
        // ---- stage 1: A^T[key,q] = sum_d K[key,d] * Q[q,d]
        v8f a1 = vzero;
        const u16t* Kr = Kp + (size_t)(b*T + j*32  + keyoff + ln) * D;
        const u16t* Qr = Qp + (size_t)(b*T + qt*32 + qoff   + ln) * D;
#pragma unroll 4
        for (int d0 = 0; d0 < D; d0 += 32) {
            v16bf af = make_frag(*(const u16x8*)(Kr + d0 + 8*half),
                                 *(const u16x8*)(Kr + d0 + 16 + 8*half));
            v16bf bf = make_frag(*(const u16x8*)(Qr + d0 + 8*half),
                                 *(const u16x8*)(Qr + d0 + 16 + 8*half));
            a1 = wmma_bf16(af, bf, a1);
        }
        if (j == qt) {                           // causal mask on diagonal tile
            const int q_rel = qoff + ln;
#pragma unroll
            for (int v = 0; v < 8; ++v)
                if (keyoff + v + 8*half > q_rel) a1[v] = 0.f;
        }
        float part = 0.f;                        // den contribution: sum over keys
#pragma unroll
        for (int v = 0; v < 8; ++v) part += a1[v];
        atomicAdd(&den_s[qoff + ln], part);      // ds_add_f32
        u16x8 pk;                                // 8 consecutive keys per lane
#pragma unroll
        for (int v = 0; v < 8; ++v) pk[v] = f2bf(a1[v]);
        *(u16x8*)(&As[qoff + ln][keyoff + 8*half]) = pk;   // ds_store_b128
        __syncthreads();

        // ---- stage 2: num[q,col] += A[q,key] * Vt[col,key]^T
        v16bf aa[2];
#pragma unroll
        for (int m = 0; m < 2; ++m) {
            const u16t* ap = &As[m*16 + ln][0];
            aa[m] = make_frag(*(const u16x8*)(ap + 8*half),
                              *(const u16x8*)(ap + 16 + 8*half));
        }
        const u16t* vb = Vt + (((size_t)(b*1024)) << 11) + j*32 + 8*half;
#pragma unroll
        for (int n = 0; n < 8; ++n) {
            const u16t* vp = vb + (((size_t)(colbase + n*16 + ln)) << 11);
            v16bf bf = make_frag(*(const u16x8*)vp, *(const u16x8*)(vp + 16));
#pragma unroll
            for (int m = 0; m < 2; ++m) acc[m][n] = wmma_bf16(aa[m], bf, acc[m][n]);
        }
        __syncthreads();
    }

    // ---- epilogue: O = num / (den + 1e-6) -> bf16 row-major for output GEMM
#pragma unroll
    for (int m = 0; m < 2; ++m) {
#pragma unroll
        for (int v = 0; v < 8; ++v) {
            const int q_rel = m*16 + v + 8*half;
            const float inv = 1.0f / (den_s[q_rel] + 1e-6f);
            const int t = qt*32 + q_rel;
#pragma unroll
            for (int n = 0; n < 8; ++n) {
                const int col = colbase + n*16 + ln;
                Og[(size_t)(b*T + t) * D + col] = f2bf(acc[m][n][v] * inv);
            }
        }
    }
}

// ---------------------------------------------------------------------------
extern "C" void kernel_launch(void* const* d_in, const int* in_sizes, int n_in,
                              void* d_out, int out_size, void* d_ws, size_t ws_size,
                              hipStream_t stream) {
    const float* x  = (const float*)d_in[0];
    const float* Wq = (const float*)d_in[1];
    const float* bq = (const float*)d_in[2];
    const float* Wk = (const float*)d_in[3];
    const float* bk = (const float*)d_in[4];
    const float* Wv = (const float*)d_in[5];
    const float* bv = (const float*)d_in[6];
    const float* Wo = (const float*)d_in[7];
    const float* bo = (const float*)d_in[8];

    const int B = 4, T = 2048, D = 1024, M = B * T;  // M = 8192
    char* ws = (char*)d_ws;
    u16t* xb  = (u16t*)ws;                                   // 16 MB
    u16t* Wqb = (u16t*)(ws + (size_t)16*1024*1024);          // 2 MB each
    u16t* Wkb = Wqb + (size_t)D*D;
    u16t* Wvb = Wkb + (size_t)D*D;
    u16t* Wob = Wvb + (size_t)D*D;
    u16t* Qp  = (u16t*)(ws + (size_t)24*1024*1024);          // 16 MB each
    u16t* Kp  = Qp + (size_t)M*D;
    u16t* Vt  = Kp + (size_t)M*D;   // transposed: [b][d][t]
    u16t* Og  = Vt + (size_t)M*D;

    const int nx = M * D, nw = D * D;
    cvt_f32_bf16<<<(nx + 255) / 256, 256, 0, stream>>>(x,  xb,  nx);
    cvt_f32_bf16<<<(nw + 255) / 256, 256, 0, stream>>>(Wq, Wqb, nw);
    cvt_f32_bf16<<<(nw + 255) / 256, 256, 0, stream>>>(Wk, Wkb, nw);
    cvt_f32_bf16<<<(nw + 255) / 256, 256, 0, stream>>>(Wv, Wvb, nw);
    cvt_f32_bf16<<<(nw + 255) / 256, 256, 0, stream>>>(Wo, Wob, nw);

    dim3 pg(M / 128, D / 64);
    proj_kernel<<<pg, 128, 0, stream>>>(xb, Wqb, bq, Qp, nullptr, 0);  // phi(xWq+bq)
    proj_kernel<<<pg, 128, 0, stream>>>(xb, Wkb, bk, Kp, nullptr, 0);  // phi(xWk+bk)
    proj_kernel<<<pg, 128, 0, stream>>>(xb, Wvb, bv, Vt, nullptr, 2);  // (xWv+bv)^T

    attn_kernel<<<dim3(T / 32, B, 2), 128, 0, stream>>>(Qp, Kp, Vt, Og);

    proj_kernel<<<pg, 128, 0, stream>>>(Og, Wob, bo, nullptr, (float*)d_out, 3);
}
